// QuantizedLinear_81381040324611
// MI455X (gfx1250) — matrix operand order repaired and checked by reading, not compiled, not run on hardware
//
#include <hip/hip_runtime.h>

// ---------------------------------------------------------------------------
// Quantized linear  y = x @ (S*(q - Z))^T + b   on MI455X (gfx1250, wave32)
//
// M=8192 (tokens), N=16384 (out), K=4096 (in).  Compute-bound at 16-bit WMMA
// (1.1 TFLOP vs ~0.9 GB min traffic). q-Z is an exact f16 integer, so prepack
// W to f16 once (134 MB, fits 192MB L2), prepack x to f16 (67 MB), then run an
// f16 WMMA GEMM with f32 accumulation. Epilogue: y = S*acc + bias.
//
// GEMM: 256 threads (8 waves), block tile 128x256, wave tile 64x64
//       (4x4 subtiles of v_wmma_f32_16x16x32_f16), KB=32 per step.
//       Double-buffered LDS fed by BOTH CDNA5 async paths:
//         - A tile: per-lane GLOBAL_LOAD_ASYNC_TO_LDS_B128 (ASYNCcnt)
//         - B tile: TENSOR_LOAD_TO_LDS (TDM, TENSORcnt), one descriptor per
//           k-step issued by wave 0; D# pad fields reproduce the padded
//           LDS row stride (16 DWORDs data + 4 DWORDs pad = LDP=40 f16).
//       Explicit s_wait_asynccnt / s_wait_tensorcnt before each barrier.
//
// Workspace use: W_f16 (OUT_F*IN_F*2 B) then X_f16 (TOKENS*IN_F*2 B) = 192 MB.
// ---------------------------------------------------------------------------

typedef _Float16 v8h  __attribute__((ext_vector_type(8)));
typedef _Float16 v16h __attribute__((ext_vector_type(16)));
typedef float    v8f  __attribute__((ext_vector_type(8)));
typedef unsigned int u32x4 __attribute__((ext_vector_type(4)));
typedef int          i32x4 __attribute__((ext_vector_type(4)));
typedef int          i32x8 __attribute__((ext_vector_type(8)));

constexpr int IN_F   = 4096;
constexpr int OUT_F  = 16384;
constexpr int TOKENS = 8192;

constexpr int BM = 128;        // block tile rows (tokens)
constexpr int BN = 256;        // block tile cols (out features)
constexpr int KB = 32;         // K elements per step (one wmma K)
constexpr int LDP = 40;        // padded LDS row stride in f16 (80B = 20 banks)
constexpr int THREADS = 256;   // 8 wave32 waves

// ---- prepack: int32 q -> f16 (q - Z), exact for q in [0,256), integral Z ----
__global__ void prepack_w_kernel(const int* __restrict__ q,
                                 const float* __restrict__ zp,
                                 _Float16* __restrict__ w) {
  const float Z = zp[0];
  const size_t base = ((size_t)blockIdx.x * THREADS + threadIdx.x) * 8;
  const int4 a = *(const int4*)(q + base);
  const int4 b = *(const int4*)(q + base + 4);
  v8h o;
  o[0] = (_Float16)((float)a.x - Z); o[1] = (_Float16)((float)a.y - Z);
  o[2] = (_Float16)((float)a.z - Z); o[3] = (_Float16)((float)a.w - Z);
  o[4] = (_Float16)((float)b.x - Z); o[5] = (_Float16)((float)b.y - Z);
  o[6] = (_Float16)((float)b.z - Z); o[7] = (_Float16)((float)b.w - Z);
  *(v8h*)(w + base) = o;
}

// ---- prepack: f32 x -> f16 ----
__global__ void prepack_x_kernel(const float* __restrict__ x,
                                 _Float16* __restrict__ xh) {
  const size_t base = ((size_t)blockIdx.x * THREADS + threadIdx.x) * 8;
  const float4 a = *(const float4*)(x + base);
  const float4 b = *(const float4*)(x + base + 4);
  v8h o;
  o[0] = (_Float16)a.x; o[1] = (_Float16)a.y;
  o[2] = (_Float16)a.z; o[3] = (_Float16)a.w;
  o[4] = (_Float16)b.x; o[5] = (_Float16)b.y;
  o[6] = (_Float16)b.z; o[7] = (_Float16)b.w;
  *(v8h*)(xh + base) = o;
}

// ---- main GEMM: C[M,N] = A[M,K] * B[N,K]^T, f16 in, f32 accum ----
__global__ __launch_bounds__(THREADS)
void qgemm_kernel(const _Float16* __restrict__ A,   // [TOKENS][IN_F] f16
                  const _Float16* __restrict__ B,   // [OUT_F][IN_F]  f16 (q - Z)
                  const float* __restrict__ scale,  // scalar S
                  const float* __restrict__ bias,   // [OUT_F]
                  float* __restrict__ out) {        // [TOKENS][OUT_F] f32
  __shared__ _Float16 sm[2][(BM + BN) * LDP];       // A tile then B tile, 60 KB

  const int tid  = threadIdx.x;
  const int lane = tid & 31;
  const int wave = tid >> 5;
  const int wm   = wave & 1;   // 2 waves over M
  const int wn   = wave >> 1;  // 4 waves over N
  const int m16  = lane & 15;
  const int hi   = lane >> 4;

  const int gm0 = blockIdx.y * BM;
  const int gn0 = blockIdx.x * BN;

  v8f acc[4][4] = {};

  const unsigned ldsBase = (unsigned)(uintptr_t)&sm[0][0];  // LDS byte addr
  const unsigned BUFB = (unsigned)((BM + BN) * LDP * 2);    // 30720 B pitch

  // --- A tile: per-lane async copies (GVS: mem = SGPR base + VGPR offset) ---
  unsigned aOffG[2], aLds[2];   // 2 x 16B chunks of the A tile per thread
#pragma unroll
  for (int u = 0; u < 2; ++u) {
    const int c = tid + u * THREADS;              // [0,512): 128 rows x 4 segs
    const int row = c >> 2, seg = c & 3;
    aOffG[u] = (unsigned)(((gm0 + row) * IN_F + seg * 8) * 2);
    aLds[u]  = ldsBase + (unsigned)((row * LDP + seg * 8) * 2);
  }
  auto acopyA = [&](int ks, int buf) {
    const unsigned long long ab = (unsigned long long)A + (unsigned)(ks * KB * 2);
    const unsigned bo = buf ? BUFB : 0u;
#pragma unroll
    for (int u = 0; u < 2; ++u)
      asm volatile("global_load_async_to_lds_b128 %0, %1, %2"
                   :: "v"(aLds[u] + bo), "v"(aOffG[u]), "s"(ab) : "memory");
  };

  // --- B tile: one TDM descriptor per k-step (wave 0 only; EXEC-ignored,
  //     tracked by wave 0's TENSORcnt). 2D tile: 256 rows x 32 f16, row
  //     stride IN_F, LDS padding 4 DWORDs every 16 DWORDs -> LDP=40 f16. ---
  auto tdmB = [&](int ks, int buf) {
    const unsigned long long ga =
        (unsigned long long)B + (unsigned)((gn0 * IN_F + ks * KB) * 2);
    const unsigned lds = ldsBase + (buf ? BUFB : 0u) + (unsigned)(BM * LDP * 2);
    u32x4 g0;
    g0[0] = 1u;                                   // count=1, user mode, no gather
    g0[1] = lds;                                  // lds_addr
    g0[2] = (unsigned)ga;                         // global_addr[31:0]
    g0[3] = (unsigned)(ga >> 32) | (2u << 30);    // global_addr[56:32] | type=2
    i32x8 g1;
    g1[0] = 0x06D10000;          // data_size=2B | pad_en | pad_int=16dw | pad_amt=4dw
    g1[1] = (int)((IN_F & 0xFFFF) << 16);         // tensor_dim0 = 4096 (lo16 here)
    g1[2] = (int)((unsigned)BN << 16);            // tensor_dim0 hi=0 | tensor_dim1=256
    g1[3] = (int)((unsigned)KB << 16);            // tensor_dim1 hi=0 | tile_dim0=32
    g1[4] = BN;                                   // tile_dim1=256 | tile_dim2=0
    g1[5] = IN_F;                                 // tensor_dim0_stride = 4096
    g1[6] = 0;                                    // stride hi | dim1_stride lo
    g1[7] = 0;
    const i32x4 z4 = {0, 0, 0, 0};                // groups 2/3: dims 3/4 unused
    const i32x8 z8 = {0, 0, 0, 0, 0, 0, 0, 0};    // 6-arg toolchain: extra group
    __builtin_amdgcn_tensor_load_to_lds(g0, g1, z4, z4, z8, 0);
  };

  acopyA(0, 0);
  if (wave == 0) tdmB(0, 0);
  asm volatile("s_wait_asynccnt 0" ::: "memory");
  if (wave == 0) __builtin_amdgcn_s_wait_tensorcnt(0);
  __syncthreads();

  constexpr int KSTEPS = IN_F / KB;  // 128
  for (int ks = 0; ks < KSTEPS; ++ks) {
    const int buf = ks & 1;
    // Fill the other buffer while we compute (its last readers all passed
    // the barrier that ended iteration ks-1).
    if (ks + 1 < KSTEPS) {
      acopyA(ks + 1, buf ^ 1);
      if (wave == 0) tdmB(ks + 1, buf ^ 1);
    }

    v16h af[4], bf[4];
    // A fragment (16x32 f16, ISA 7.12.2): lane hi=0 -> K{0..7,16..23},
    // hi=1 -> K{8..15,24..31}; row = m16.
#pragma unroll
    for (int i = 0; i < 4; ++i) {
      const _Float16* p = &sm[buf][(wm * 64 + i * 16 + m16) * LDP];
      v8h lo = *(const v8h*)(p + hi * 8);
      v8h hh = *(const v8h*)(p + 16 + hi * 8);
      af[i] = __builtin_shufflevector(lo, hh, 0,1,2,3,4,5,6,7,8,9,10,11,12,13,14,15);
    }
    // B fragment (32x16): lane n = m16 holds K{hi*8..hi*8+7, 16+hi*8..16+hi*8+7}
    // (column n of B == row n of W, both K-contiguous -> no transpose needed).
#pragma unroll
    for (int j = 0; j < 4; ++j) {
      const _Float16* p = &sm[buf][BM * LDP + (wn * 64 + j * 16 + m16) * LDP];
      v8h lo = *(const v8h*)(p + hi * 8);
      v8h hh = *(const v8h*)(p + 16 + hi * 8);
      bf[j] = __builtin_shufflevector(lo, hh, 0,1,2,3,4,5,6,7,8,9,10,11,12,13,14,15);
    }

#pragma unroll
    for (int i = 0; i < 4; ++i)
#pragma unroll
      for (int j = 0; j < 4; ++j)
        acc[i][j] = __builtin_amdgcn_wmma_f32_16x16x32_f16(
            false, af[i], false, bf[j], (short)0, acc[i][j], false, false);

    if (ks + 1 < KSTEPS) {
      asm volatile("s_wait_asynccnt 0" ::: "memory");   // our A-chunks landed
      if (wave == 0) __builtin_amdgcn_s_wait_tensorcnt(0);  // B tile landed
      __syncthreads();                                  // publish to all waves
    }
  }

  // Epilogue: y = S*acc + bias. C/D layout: VGPR r holds M=r (lanes 0-15,N=lane)
  // and M=r+8 (lanes 16-31, N=lane-16).
  const float S = scale[0];
#pragma unroll
  for (int j = 0; j < 4; ++j) {
    const int col = gn0 + wn * 64 + j * 16 + m16;
    const float bb = bias[col];
#pragma unroll
    for (int i = 0; i < 4; ++i) {
      const int row0 = gm0 + wm * 64 + i * 16 + hi * 8;
#pragma unroll
      for (int r = 0; r < 8; ++r)
        out[(size_t)(row0 + r) * OUT_F + col] = S * acc[i][j][r] + bb;
    }
  }
}

extern "C" void kernel_launch(void* const* d_in, const int* in_sizes, int n_in,
                              void* d_out, int out_size, void* d_ws, size_t ws_size,
                              hipStream_t stream) {
  const float* x     = (const float*)d_in[0];
  const int*   qw    = (const int*)d_in[1];
  const float* s     = (const float*)d_in[2];
  const float* zp    = (const float*)d_in[3];
  const float* bias  = (const float*)d_in[4];
  float*       out   = (float*)d_out;

  _Float16* wf16 = (_Float16*)d_ws;                       // 134,217,728 B
  _Float16* xf16 = wf16 + (size_t)OUT_F * IN_F;           // +67,108,864 B

  {
    const int total = OUT_F * IN_F;                       // 67,108,864
    prepack_w_kernel<<<total / (8 * THREADS), THREADS, 0, stream>>>(qw, zp, wf16);
  }
  {
    const int total = TOKENS * IN_F;                      // 33,554,432
    prepack_x_kernel<<<total / (8 * THREADS), THREADS, 0, stream>>>(x, xf16);
  }
  {
    dim3 grid(OUT_F / BN, TOKENS / BM);                   // 64 x 64 blocks
    qgemm_kernel<<<grid, THREADS, 0, stream>>>(xf16, wf16, s, bias, out);
  }
}
